// CrossMultiHeadAttention_62723702391701
// MI455X (gfx1250) — compile-verified, matrix operand order checked by reading, not run on hardware
//
#include <hip/hip_runtime.h>
#include <hip/hip_bf16.h>

// ---------------------------------------------------------------------------
// CDNA5 (gfx1250) cross multi-head attention
//   B=4, S=T=2048, E=1024, H=16, D=64
//   fp32 inputs -> bf16 compute with f32 WMMA accumulation
// All B-operands (weights, V) are kept pre-transposed so every LDS staging
// operation is a 128-bit copy (no scalar scatter transposes in hot loops).
// ---------------------------------------------------------------------------

typedef __attribute__((ext_vector_type(16))) __bf16          v16bf;
typedef __attribute__((ext_vector_type(16))) unsigned short  v16u;
typedef __attribute__((ext_vector_type(8)))  unsigned short  v8u;
typedef __attribute__((ext_vector_type(8)))  float           v8f;

union Frag16 {
    v16bf bf;
    v16u  u;
    v8u   h[2];
};

__device__ __forceinline__ unsigned short f32_to_bf16(float f) {
    union { float f; unsigned int u; } v;
    v.f = f;
    unsigned int r = v.u + 0x7FFFu + ((v.u >> 16) & 1u);  // round-to-nearest-even
    return (unsigned short)(r >> 16);
}

// ---------------------------------------------------------------------------
// fp32 -> bf16 elementwise convert
// ---------------------------------------------------------------------------
__global__ void cvt_f32_bf16_kernel(const float* __restrict__ src,
                                    unsigned short* __restrict__ dst, int n) {
    int i = blockIdx.x * blockDim.x + threadIdx.x;
    if (i < n) dst[i] = f32_to_bf16(src[i]);
}

// ---------------------------------------------------------------------------
// W[h][e][d] (16,1024,64) -> Wt[h*64+d][e] (1024,1024), fp32 -> bf16
// (transposed concat so GEMM B-fragments are contiguous rows)
// ---------------------------------------------------------------------------
__global__ void repack_w_t_kernel(const float* __restrict__ w,
                                  unsigned short* __restrict__ dst) {
    int o = blockIdx.x * blockDim.x + threadIdx.x;  // < 1024*1024, dst[n][e]
    int n = o >> 10;
    int e = o & 1023;
    int h = n >> 6;
    int d = n & 63;
    dst[o] = f32_to_bf16(w[(h << 16) + (e << 6) + d]);  // h*1024*64 + e*64 + d
}

// ---------------------------------------------------------------------------
// Wo[k][n] (1024,1024) -> Wot[n][k], fp32 -> bf16
// ---------------------------------------------------------------------------
__global__ void transpose_cvt_kernel(const float* __restrict__ src,
                                     unsigned short* __restrict__ dst) {
    int o = blockIdx.x * blockDim.x + threadIdx.x;  // < 1024*1024, dst[n][k]
    int n = o >> 10;
    int k = o & 1023;
    dst[o] = f32_to_bf16(src[(k << 10) + n]);
}

// ---------------------------------------------------------------------------
// Tiled bf16 GEMM with f32 WMMA accumulation:
//   C[M,N] = A[M,K] * Bt[N,K]^T + bias[N]      (B supplied pre-transposed)
// Block tile 128(M) x 128(N), KT=32.  8 waves: 4(M) x 2(N), wave tile 32x64.
// OUT_MODE: 0 -> bf16 [M,N], 1 -> f32 [M,N], 2 -> bf16 transposed [N,M].
// ---------------------------------------------------------------------------
template <int OUT_MODE>
__global__ __launch_bounds__(256)
void gemm_bf16_kernel(const unsigned short* __restrict__ A,
                      const unsigned short* __restrict__ Bt,
                      const float* __restrict__ bias,
                      void* __restrict__ Cout,
                      int M, int N, int K) {
    constexpr int BM = 128, BN = 128, KT = 32;
    __shared__ unsigned short As[BM * KT];   // [128][32] (m, k)
    __shared__ unsigned short Bs[BN * KT];   // [128][32] (n, k)

    const int tid  = threadIdx.x;
    const int wave = tid >> 5;
    const int lane = tid & 31;
    const int lrow = lane & 15;
    const int lhi  = lane >> 4;
    const int wm   = wave >> 1;  // 0..3 : M strip of 32
    const int wn   = wave & 1;   // 0..1 : N strip of 64
    const int m0   = blockIdx.x * BM;
    const int n0   = blockIdx.y * BN;

    v8f acc[2][4];
#pragma unroll
    for (int i = 0; i < 2; i++)
#pragma unroll
        for (int j = 0; j < 4; j++)
#pragma unroll
            for (int e = 0; e < 8; e++) acc[i][j][e] = 0.0f;

    const int srow = tid >> 1;              // 0..127
    const int scol = (tid & 1) << 4;        // 0 or 16

    for (int k0 = 0; k0 < K; k0 += KT) {
        // ---- stage A tile 128x32 and B tile 128x32, pure 128-bit copies ----
        {
            const unsigned short* ga = A + (size_t)(m0 + srow) * K + k0 + scol;
            *(v8u*)(As + srow * KT + scol)     = *(const v8u*)(ga);
            *(v8u*)(As + srow * KT + scol + 8) = *(const v8u*)(ga + 8);
            const unsigned short* gb = Bt + (size_t)(n0 + srow) * K + k0 + scol;
            *(v8u*)(Bs + srow * KT + scol)     = *(const v8u*)(gb);
            *(v8u*)(Bs + srow * KT + scol + 8) = *(const v8u*)(gb + 8);
        }
        __syncthreads();

        // ---- fragments: 16-bit layout, K = (e/8)*16 + (lane/16)*8 + e%8 ----
        Frag16 af[2], bfg[4];
#pragma unroll
        for (int i = 0; i < 2; i++) {
            int r = wm * 32 + i * 16 + lrow;
            af[i].h[0] = *(const v8u*)(As + r * KT + lhi * 8);
            af[i].h[1] = *(const v8u*)(As + r * KT + 16 + lhi * 8);
        }
#pragma unroll
        for (int j = 0; j < 4; j++) {
            int c = wn * 64 + j * 16 + lrow;
            bfg[j].h[0] = *(const v8u*)(Bs + c * KT + lhi * 8);
            bfg[j].h[1] = *(const v8u*)(Bs + c * KT + 16 + lhi * 8);
        }
#pragma unroll
        for (int i = 0; i < 2; i++)
#pragma unroll
            for (int j = 0; j < 4; j++)
                acc[i][j] = __builtin_amdgcn_wmma_f32_16x16x32_bf16(
                    false, af[i].bf, false, bfg[j].bf, (short)0, acc[i][j],
                    false, false);
        __syncthreads();
    }

    // ---- epilogue: D layout: VGPR r -> row r + 8*(lane/16), col = lane%16 ----
#pragma unroll
    for (int i = 0; i < 2; i++)
#pragma unroll
        for (int j = 0; j < 4; j++) {
            int col = n0 + wn * 64 + j * 16 + lrow;
            float bv = bias[col];
#pragma unroll
            for (int r = 0; r < 8; r++) {
                int rowg = m0 + wm * 32 + i * 16 + r + lhi * 8;
                float vv = acc[i][j][r] + bv;
                if (OUT_MODE == 1)
                    ((float*)Cout)[(size_t)rowg * N + col] = vv;
                else if (OUT_MODE == 2)   // transposed store: C^T[n][m]
                    ((unsigned short*)Cout)[(size_t)col * M + rowg] =
                        f32_to_bf16(vv);
                else
                    ((unsigned short*)Cout)[(size_t)rowg * N + col] =
                        f32_to_bf16(vv);
            }
        }
}

// ---------------------------------------------------------------------------
// Flash attention (online softmax), bf16 Q/K/V, f32 accumulation.
// Grid: (S/128, H, B).  256 threads = 8 waves; each wave owns 16 Q rows.
// Q,K are [B*S, 1024] (head h at cols h*64..); V is transposed: [1024, B*T].
// ---------------------------------------------------------------------------
__global__ __launch_bounds__(256)
void flash_attn_kernel(const unsigned short* __restrict__ Q,
                       const unsigned short* __restrict__ Kb,
                       const unsigned short* __restrict__ Vt,
                       unsigned short* __restrict__ O,
                       int S, int T, int MS) {
    constexpr int HD = 64;
    const int h    = blockIdx.y;
    const int b    = blockIdx.z;
    const int tid  = threadIdx.x;
    const int wave = tid >> 5;
    const int lane = tid & 31;
    const int lrow = lane & 15;
    const int lhi  = lane >> 4;
    const int s0   = blockIdx.x * 128 + wave * 16;

    __shared__ unsigned short Ks[64 * 64];      // [t][d]
    __shared__ unsigned short Vs[64 * 64];      // [d][t]
    __shared__ unsigned short Ps[8][16 * 64];   // per-wave P staging [s][t]

    // ---- load Q fragments once: A-matrix 16x64 split in two K=32 chunks ----
    const size_t rowQ = (size_t)(b * S + s0 + lrow) * 1024 + h * HD;
    Frag16 qf[2];
#pragma unroll
    for (int kc = 0; kc < 2; kc++)
#pragma unroll
        for (int hh = 0; hh < 2; hh++)
            qf[kc].h[hh] = *(const v8u*)(Q + rowQ + kc * 32 + hh * 16 + lhi * 8);

    v8f oacc[4];
#pragma unroll
    for (int j = 0; j < 4; j++)
#pragma unroll
        for (int e = 0; e < 8; e++) oacc[j][e] = 0.0f;

    float mrow[8], lsum[8];
#pragma unroll
    for (int r = 0; r < 8; r++) { mrow[r] = -1.0e30f; lsum[r] = 0.0f; }

    const float scale = 0.125f;  // 1/sqrt(64)
    const int strow = tid >> 2;           // 0..63
    const int stcol = (tid & 3) << 4;     // 0,16,32,48

    for (int t0 = 0; t0 < T; t0 += 64) {
        // ---- cooperative tile staging, pure 128-bit copies ----
        {
            // K tile: Ks[t][d] from K rows
            const unsigned short* gk =
                Kb + (size_t)(b * T + t0 + strow) * 1024 + h * HD + stcol;
            *(v8u*)(Ks + strow * 64 + stcol)     = *(const v8u*)(gk);
            *(v8u*)(Ks + strow * 64 + stcol + 8) = *(const v8u*)(gk + 8);
            // V tile: Vs[d][t] from V^T rows
            const unsigned short* gv =
                Vt + (size_t)(h * HD + strow) * MS + b * T + t0 + stcol;
            *(v8u*)(Vs + strow * 64 + stcol)     = *(const v8u*)(gv);
            *(v8u*)(Vs + strow * 64 + stcol + 8) = *(const v8u*)(gv + 8);
        }
        __syncthreads();

        // ---- scores: 16 x 64 tile = 4 WMMA column tiles, K=64 over d ----
        v8f sc[4];
#pragma unroll
        for (int j = 0; j < 4; j++) {
#pragma unroll
            for (int e = 0; e < 8; e++) sc[j][e] = 0.0f;
#pragma unroll
            for (int kc = 0; kc < 2; kc++) {
                Frag16 kf;
                int trow = j * 16 + lrow;
                kf.h[0] = *(const v8u*)(Ks + trow * 64 + kc * 32 + lhi * 8);
                kf.h[1] = *(const v8u*)(Ks + trow * 64 + kc * 32 + 16 + lhi * 8);
                sc[j] = __builtin_amdgcn_wmma_f32_16x16x32_bf16(
                    false, qf[kc].bf, false, kf.bf, (short)0, sc[j], false, false);
            }
#pragma unroll
            for (int e = 0; e < 8; e++) sc[j][e] *= scale;
        }

        // ---- online softmax (stats in D layout: VGPR r -> row r+8*lhi) ----
        float mnew[8], alpha[8];
#pragma unroll
        for (int r = 0; r < 8; r++) {
            float tm = fmaxf(fmaxf(sc[0][r], sc[1][r]), fmaxf(sc[2][r], sc[3][r]));
#pragma unroll
            for (int msk = 1; msk < 16; msk <<= 1)
                tm = fmaxf(tm, __shfl_xor(tm, msk, 32));
            mnew[r]  = fmaxf(mrow[r], tm);
            alpha[r] = __expf(mrow[r] - mnew[r]);
            mrow[r]  = mnew[r];
        }
#pragma unroll
        for (int j = 0; j < 4; j++)
#pragma unroll
            for (int r = 0; r < 8; r++)
                sc[j][r] = __expf(sc[j][r] - mnew[r]);
#pragma unroll
        for (int r = 0; r < 8; r++) {
            float s = (sc[0][r] + sc[1][r]) + (sc[2][r] + sc[3][r]);
#pragma unroll
            for (int msk = 1; msk < 16; msk <<= 1)
                s += __shfl_xor(s, msk, 32);
            lsum[r] = lsum[r] * alpha[r] + s;
        }
#pragma unroll
        for (int j = 0; j < 4; j++)
#pragma unroll
            for (int r = 0; r < 8; r++) oacc[j][r] *= alpha[r];

        // ---- re-shape P: D layout -> A layout via per-wave LDS staging ----
        unsigned short* pw = &Ps[wave][0];
#pragma unroll
        for (int j = 0; j < 4; j++)
#pragma unroll
            for (int r = 0; r < 8; r++)
                pw[(r + lhi * 8) * 64 + j * 16 + lrow] = f32_to_bf16(sc[j][r]);
        // wave-private LDS round-trip: wait on DS counter only
        asm volatile("s_wait_dscnt 0x0" ::: "memory");

        // ---- O += P @ V : 4 d-tiles, K=64 over t ----
#pragma unroll
        for (int kc = 0; kc < 2; kc++) {
            Frag16 pf;
            pf.h[0] = *(const v8u*)(pw + lrow * 64 + kc * 32 + lhi * 8);
            pf.h[1] = *(const v8u*)(pw + lrow * 64 + kc * 32 + 16 + lhi * 8);
#pragma unroll
            for (int jd = 0; jd < 4; jd++) {
                Frag16 vf;
                int dcol = jd * 16 + lrow;
                vf.h[0] = *(const v8u*)(Vs + dcol * 64 + kc * 32 + lhi * 8);
                vf.h[1] = *(const v8u*)(Vs + dcol * 64 + kc * 32 + 16 + lhi * 8);
                oacc[jd] = __builtin_amdgcn_wmma_f32_16x16x32_bf16(
                    false, pf.bf, false, vf.bf, (short)0, oacc[jd], false, false);
            }
        }
        __syncthreads();
    }

    // ---- finalize: O /= l, store bf16 to concat layout [B*S][H*64] ----
#pragma unroll
    for (int r = 0; r < 8; r++) lsum[r] = 1.0f / lsum[r];
#pragma unroll
    for (int jd = 0; jd < 4; jd++) {
        int dcol = h * HD + jd * 16 + lrow;
#pragma unroll
        for (int r = 0; r < 8; r++) {
            size_t idx = (size_t)(b * S + s0 + r + lhi * 8) * 1024 + dcol;
            O[idx] = f32_to_bf16(oacc[jd][r] * lsum[r]);
        }
    }
}

// ---------------------------------------------------------------------------
// host-side orchestration
// ---------------------------------------------------------------------------
extern "C" void kernel_launch(void* const* d_in, const int* in_sizes, int n_in,
                              void* d_out, int out_size, void* d_ws, size_t ws_size,
                              hipStream_t stream) {
    (void)in_sizes; (void)n_in; (void)out_size; (void)ws_size;

    const float* x  = (const float*)d_in[0];
    const float* y  = (const float*)d_in[1];
    const float* Wq = (const float*)d_in[2];
    const float* bq = (const float*)d_in[3];
    const float* Wk = (const float*)d_in[4];
    const float* bk = (const float*)d_in[5];
    const float* Wv = (const float*)d_in[6];
    const float* bv = (const float*)d_in[7];
    const float* Wo = (const float*)d_in[8];
    const float* bo = (const float*)d_in[9];
    float* out = (float*)d_out;

    const int B = 4, S = 2048, T = 2048, E = 1024, H = 16;
    const int MS = B * S;  // 8192 flattened rows

    // workspace layout (bf16 as u16); total ~104 MB
    unsigned short* ws  = (unsigned short*)d_ws;
    size_t off = 0;
    unsigned short* xb  = ws + off; off += (size_t)MS * E;
    unsigned short* yb  = ws + off; off += (size_t)MS * E;
    unsigned short* wqb = ws + off; off += (size_t)E * E;  // [n][k] transposed
    unsigned short* wkb = ws + off; off += (size_t)E * E;
    unsigned short* wvb = ws + off; off += (size_t)E * E;
    unsigned short* wob = ws + off; off += (size_t)E * E;  // Wo^T
    unsigned short* Qb  = ws + off; off += (size_t)MS * E; // [m][n]
    unsigned short* Kb  = ws + off; off += (size_t)MS * E; // [m][n]
    unsigned short* Vtb = ws + off; off += (size_t)MS * E; // [n][m] (V^T)
    unsigned short* Ob  = ws + off; off += (size_t)MS * E; // [m][n]

    // 1) converts / repacks (one-shot, tiny vs GEMM cost)
    {
        int n = MS * E;
        cvt_f32_bf16_kernel<<<(n + 255) / 256, 256, 0, stream>>>(x, xb, n);
        cvt_f32_bf16_kernel<<<(n + 255) / 256, 256, 0, stream>>>(y, yb, n);
        int nw = E * E;
        repack_w_t_kernel<<<(nw + 255) / 256, 256, 0, stream>>>(Wq, wqb);
        repack_w_t_kernel<<<(nw + 255) / 256, 256, 0, stream>>>(Wk, wkb);
        repack_w_t_kernel<<<(nw + 255) / 256, 256, 0, stream>>>(Wv, wvb);
        transpose_cvt_kernel<<<(nw + 255) / 256, 256, 0, stream>>>(Wo, wob);
    }

    // 2) Q/K/V projections: [8192,1024] x [1024,1024]^T + bias
    dim3 gg(MS / 128, E / 128);
    gemm_bf16_kernel<0><<<gg, 256, 0, stream>>>(xb, wqb, bq, Qb, MS, E, E);
    gemm_bf16_kernel<0><<<gg, 256, 0, stream>>>(yb, wkb, bk, Kb, MS, E, E);
    gemm_bf16_kernel<2><<<gg, 256, 0, stream>>>(yb, wvb, bv, Vtb, MS, E, E);

    // 3) flash attention per (b, h, 128-row tile)
    dim3 fg(S / 128, H, B);
    flash_attn_kernel<<<fg, 256, 0, stream>>>(Qb, Kb, Vtb, Ob, S, T, MS);

    // 4) output projection -> fp32 d_out
    gemm_bf16_kernel<1><<<gg, 256, 0, stream>>>(Ob, wob, bo, out, MS, E, E);
}
